// PhysiologicalGNN_67568425501221
// MI455X (gfx1250) — compile-verified, matrix operand order checked by reading, not run on hardware
//
#include <hip/hip_runtime.h>
#include <stdint.h>

// ---------------- problem constants ----------------
static constexpr int NN   = 30720;       // nodes
static constexpr int NE   = 600000;      // edges
static constexpr int NET  = NE + NN;     // edges + self loops
static constexpr int HIDC = 64;
static constexpr int NG   = 2048;        // graphs

// ---------------- workspace layout (bytes) ----------------
static constexpr size_t AL(size_t x) { return (x + 255) & ~size_t(255); }
static constexpr size_t OFF_DF   = 0;                                    // [2048*64] f32
static constexpr size_t OFF_XC   = AL(OFF_DF  + size_t(NG)*HIDC*4);      // [N*64] f32
static constexpr size_t OFF_XB   = AL(OFF_XC  + size_t(NN)*HIDC*4);      // [N*256] bf16
static constexpr size_t OFF_WT   = AL(OFF_XB  + size_t(NN)*256*2);       // [256*256] bf16
static constexpr size_t OFF_H    = AL(OFF_WT  + size_t(256)*256*2);      // [N*256] f32
static constexpr size_t OFF_OUT  = AL(OFF_H   + size_t(NN)*256*4);       // [N*256] f32
static constexpr size_t OFF_AS   = AL(OFF_OUT + size_t(NN)*256*4);       // [N*4] f32
static constexpr size_t OFF_AD   = AL(OFF_AS  + size_t(NN)*4*4);
static constexpr size_t OFF_M    = AL(OFF_AD  + size_t(NN)*4*4);
static constexpr size_t OFF_S    = AL(OFF_M   + size_t(NN)*4*4);
static constexpr size_t OFF_MEAN = AL(OFF_S   + size_t(NN)*4*4);         // 2 f32 sums
static constexpr size_t OFF_Q    = AL(OFF_MEAN + 256);                   // 3*heads f32

// ---------------- types for WMMA ----------------
typedef __attribute__((ext_vector_type(16))) __bf16          v16bf;
typedef __attribute__((ext_vector_type(8)))  float           v8f;
typedef __attribute__((ext_vector_type(8)))  unsigned short  u16x8;

union ABPack { struct { u16x8 lo, hi; } s; v16bf v; };

#define DEVINL __device__ __forceinline__

DEVINL unsigned short f2bf(float f) {
  uint32_t u = __float_as_uint(f);
  u += 0x7fffu + ((u >> 16) & 1u);        // round-to-nearest-even
  return (unsigned short)(u >> 16);
}

static inline int cdiv(int a, int b) { return (a + b - 1) / b; }

// ---------------- generic fill ----------------
__global__ void fill_u32(uint32_t* p, uint32_t v, int n) {
  int i = blockIdx.x * blockDim.x + threadIdx.x;
  if (i < n) p[i] = v;
}

// ---------------- drug conditioning ----------------
__global__ void drug_df(const float* __restrict__ emb, const int* __restrict__ id,
                        const float* __restrict__ Wd, const float* __restrict__ bd,
                        float* __restrict__ df) {
  int t = blockIdx.x * blockDim.x + threadIdx.x;
  if (t >= NG * HIDC) return;
  int g = t / HIDC, c = t % HIDC;
  const float* e = emb + (size_t)id[g] * HIDC;
  float acc = bd[c];
  for (int k = 0; k < HIDC; ++k) acc += e[k] * Wd[k * HIDC + c];
  df[t] = acc;
}

__global__ void add_cond(const float* __restrict__ x, const float* __restrict__ df,
                         const int* __restrict__ batch, float* __restrict__ xc) {
  int t = blockIdx.x * blockDim.x + threadIdx.x;
  if (t >= NN * HIDC) return;
  int n = t / HIDC, c = t % HIDC;
  xc[t] = x[t] + df[(size_t)batch[n] * HIDC + c];
}

// ---------------- conversions ----------------
__global__ void cvt_bf16(const float* __restrict__ s, unsigned short* __restrict__ d, int n) {
  int i = blockIdx.x * blockDim.x + threadIdx.x;
  if (i < n) d[i] = f2bf(s[i]);
}

__global__ void w_transpose_bf16(const float* __restrict__ W, unsigned short* __restrict__ Wt,
                                 int din, int dout) {
  int t = blockIdx.x * blockDim.x + threadIdx.x;
  if (t >= din * dout) return;
  int o = t / din, i = t % din;
  Wt[t] = f2bf(W[(size_t)i * dout + o]);
}

// ---------------- edge attr mean (sum pass; divide in qcoef) ----------------
__global__ void ea_accum(const float* __restrict__ ea, float* __restrict__ meansum) {
  int e = blockIdx.x * blockDim.x + threadIdx.x;
  if (e >= NE) return;
  atomicAdd(&meansum[0], ea[2 * e]);
  atomicAdd(&meansum[1], ea[2 * e + 1]);
}

// q[h] = We[0,:].a_edge ; q[heads+h] = We[1,:].a_edge ; q[2h..] = loop (mean) alpha
__global__ void qcoef(const float* __restrict__ We, const float* __restrict__ ae,
                      const float* __restrict__ meansum, float* __restrict__ q,
                      int heads, int dout) {
  int h = threadIdx.x;
  if (h >= heads) return;
  float q0 = 0.f, q1 = 0.f;
  for (int c = 0; c < 64; ++c) {
    float a = ae[h * 64 + c];
    q0 += We[h * 64 + c] * a;
    q1 += We[dout + h * 64 + c] * a;
  }
  q[h] = q0;
  q[heads + h] = q1;
  float m0 = meansum[0] / (float)NE, m1 = meansum[1] / (float)NE;
  q[2 * heads + h] = m0 * q0 + m1 * q1;
}

// ---------------- WMMA GEMM: C[M,N] = A[M,K](bf16) * Bt[N,K](bf16)^T ----------------
// One wave computes a 16x64 strip of C: one A fragment feeds 4 WMMAs per K-step,
// and the K-loop is software-pipelined (loads for k+32 issued before WMMAs for k).
DEVINL void loadAB(ABPack& p, const unsigned short* base, int k) {
  p.s.lo = *(const u16x8*)(base + k);        // K = k + half*8 .. +7
  p.s.hi = *(const u16x8*)(base + k + 16);   // K = k+16+half*8 .. +7
}

__launch_bounds__(128)
__global__ void gemm_bf16(const unsigned short* __restrict__ A,
                          const unsigned short* __restrict__ Bt,
                          float* __restrict__ C,
                          int M, int N, int K, int numTiles) {
  int lane = threadIdx.x & 31;
  int wave = threadIdx.x >> 5;
  int tile = blockIdx.x * 4 + wave;
  if (tile >= numTiles) return;                 // wave-uniform exit (EXEC all-1s for WMMA)
  int nstr = N >> 6;                            // strips of 64 columns
  int tm   = (tile / nstr) << 4;
  int n0   = (tile % nstr) << 6;
  int half = lane >> 4;                         // K-split halves per ISA 16-bit A layout
  int r    = lane & 15;

  const unsigned short* arow = A + (size_t)(tm + r) * K + half * 8;
  const unsigned short* brow[4];
#pragma unroll
  for (int j = 0; j < 4; ++j)
    brow[j] = Bt + (size_t)(n0 + j * 16 + r) * K + half * 8;

  v8f acc0 = {}, acc1 = {}, acc2 = {}, acc3 = {};
  ABPack ac, bc[4];
  loadAB(ac, arow, 0);
#pragma unroll
  for (int j = 0; j < 4; ++j) loadAB(bc[j], brow[j], 0);

  for (int k = 32; k < K; k += 32) {
    // issue next K-step loads before consuming current fragments
    ABPack an, bn[4];
    loadAB(an, arow, k);
#pragma unroll
    for (int j = 0; j < 4; ++j) loadAB(bn[j], brow[j], k);
    if (k + 32 < K) {
      __builtin_prefetch(arow + k + 32, 0, 3);  // global_prefetch_b8 (near caches)
      __builtin_prefetch(brow[0] + k + 32, 0, 3);
    }
    acc0 = __builtin_amdgcn_wmma_f32_16x16x32_bf16(false, ac.v, false, bc[0].v,
                                                   (short)0, acc0, false, false);
    acc1 = __builtin_amdgcn_wmma_f32_16x16x32_bf16(false, ac.v, false, bc[1].v,
                                                   (short)0, acc1, false, false);
    acc2 = __builtin_amdgcn_wmma_f32_16x16x32_bf16(false, ac.v, false, bc[2].v,
                                                   (short)0, acc2, false, false);
    acc3 = __builtin_amdgcn_wmma_f32_16x16x32_bf16(false, ac.v, false, bc[3].v,
                                                   (short)0, acc3, false, false);
    ac = an;
#pragma unroll
    for (int j = 0; j < 4; ++j) bc[j] = bn[j];
  }
  acc0 = __builtin_amdgcn_wmma_f32_16x16x32_bf16(false, ac.v, false, bc[0].v,
                                                 (short)0, acc0, false, false);
  acc1 = __builtin_amdgcn_wmma_f32_16x16x32_bf16(false, ac.v, false, bc[1].v,
                                                 (short)0, acc1, false, false);
  acc2 = __builtin_amdgcn_wmma_f32_16x16x32_bf16(false, ac.v, false, bc[2].v,
                                                 (short)0, acc2, false, false);
  acc3 = __builtin_amdgcn_wmma_f32_16x16x32_bf16(false, ac.v, false, bc[3].v,
                                                 (short)0, acc3, false, false);

  // C layout per tile: VGPR i -> M = tm + i + half*8, N = ncol + (lane&15)
  float* crow = C + (size_t)(tm + half * 8) * N + n0 + r;
#pragma unroll
  for (int i = 0; i < 8; ++i) {
    crow[(size_t)i * N +  0] = acc0[i];
    crow[(size_t)i * N + 16] = acc1[i];
    crow[(size_t)i * N + 32] = acc2[i];
    crow[(size_t)i * N + 48] = acc3[i];
  }
}

// ---------------- per-node attention scalars ----------------
__global__ void attn_node(const float* __restrict__ H, const float* __restrict__ a_src,
                          const float* __restrict__ a_dst, float* __restrict__ as_,
                          float* __restrict__ ad_, int heads, int dout) {
  int t = blockIdx.x * blockDim.x + threadIdx.x;
  if (t >= NN * heads) return;
  int n = t / heads, h = t % heads;
  const float* hp = H + (size_t)n * dout + h * 64;
  float s1 = 0.f, s2 = 0.f;
  for (int c = 0; c < 64; ++c) {
    float v = hp[c];
    s1 += v * a_src[h * 64 + c];
    s2 += v * a_dst[h * 64 + c];
  }
  as_[t] = s1;
  ad_[t] = s2;
}

// ---------------- edge alpha helper (recomputed each pass; cheap) ----------------
DEVINL float edgeAlpha(int e, int h, const int* __restrict__ ei, const float* __restrict__ ea,
                       const float* __restrict__ q, const float* __restrict__ as_,
                       const float* __restrict__ ad_, int heads, int& r, int& c) {
  float eA;
  if (e < NE) {
    r = ei[e]; c = ei[NE + e];
    eA = ea[2 * e] * q[h] + ea[2 * e + 1] * q[heads + h];
  } else {
    r = c = e - NE;
    eA = q[2 * heads + h];                       // self-loop uses mean edge_attr
  }
  float a = as_[(size_t)r * heads + h] + ad_[(size_t)c * heads + h] + eA;
  return a > 0.f ? a : 0.2f * a;                 // leaky_relu(0.2)
}

DEVINL void atomicMaxF(float* addr, float v) {
  if (v >= 0.f) atomicMax((int*)addr, __float_as_int(v));
  else          atomicMin((unsigned int*)addr, __float_as_uint(v));
}

__global__ void edge_max(const int* ei, const float* ea, const float* q,
                         const float* as_, const float* ad_, float* m, int heads) {
  int t = blockIdx.x * blockDim.x + threadIdx.x;
  if (t >= NET * heads) return;
  int e = t / heads, h = t % heads, r, c;
  float a = edgeAlpha(e, h, ei, ea, q, as_, ad_, heads, r, c);
  atomicMaxF(&m[(size_t)c * heads + h], a);
}

__global__ void edge_sum(const int* ei, const float* ea, const float* q,
                         const float* as_, const float* ad_,
                         const float* m, float* s, int heads) {
  int t = blockIdx.x * blockDim.x + threadIdx.x;
  if (t >= NET * heads) return;
  int e = t / heads, h = t % heads, r, c;
  float a = edgeAlpha(e, h, ei, ea, q, as_, ad_, heads, r, c);
  float ex = __expf(a - m[(size_t)c * heads + h]);
  atomicAdd(&s[(size_t)c * heads + h], ex);
}

// one wave per edge; lanes stripe channels; f32 atomics land in L2
__launch_bounds__(128)
__global__ void edge_scatter(const int* ei, const float* ea, const float* q,
                             const float* as_, const float* ad_,
                             const float* m, const float* s,
                             const float* __restrict__ H, float* __restrict__ out,
                             int heads, int dout) {
  int lane = threadIdx.x & 31;
  int e = blockIdx.x * 4 + (threadIdx.x >> 5);
  if (e >= NET) return;
  int r = 0, c = 0;
  float att0 = 0.f, att1 = 0.f, att2 = 0.f, att3 = 0.f;
  for (int h = 0; h < heads; ++h) {
    float a  = edgeAlpha(e, h, ei, ea, q, as_, ad_, heads, r, c);
    float ex = __expf(a - m[(size_t)c * heads + h]);
    float at = ex / (s[(size_t)c * heads + h] + 1e-16f);
    if (h == 0) att0 = at; else if (h == 1) att1 = at;
    else if (h == 2) att2 = at; else att3 = at;
  }
  const float* hr = H + (size_t)r * dout;
  float* oc = out + (size_t)c * dout;
  for (int ch = lane; ch < dout; ch += 32) {
    int hd = ch >> 6;
    float at = hd == 0 ? att0 : hd == 1 ? att1 : hd == 2 ? att2 : att3;
    atomicAdd(&oc[ch], hr[ch] * at);
  }
}

// ---------------- bias + activation epilogue; also emits bf16 for next layer ----------------
__global__ void epilogue(float* __restrict__ out, const float* __restrict__ bias,
                         unsigned short* __restrict__ xb, int dout, int do_elu, int n) {
  int t = blockIdx.x * blockDim.x + threadIdx.x;
  if (t >= n) return;
  int ch = t % dout;
  float v = out[t] + bias[ch];
  if (do_elu) v = v > 0.f ? v : (__expf(v) - 1.f);
  out[t] = v;
  xb[t] = f2bf(v);
}

// ---------------- MLP head: relu(x @ W1 + b1) @ W2 + b2 ----------------
__global__ void head_k(const float* __restrict__ X, const float* __restrict__ W1,
                       const float* __restrict__ b1, const float* __restrict__ W2,
                       const float* __restrict__ b2, float* __restrict__ out) {
  int i = blockIdx.x * blockDim.x + threadIdx.x;
  if (i >= NN) return;
  const float* x = X + (size_t)i * 64;
  float acc = b2[0];
  for (int j = 0; j < 32; ++j) {
    float hv = b1[j];
    for (int k = 0; k < 64; ++k) hv += x[k] * W1[k * 32 + j];
    hv = hv > 0.f ? hv : 0.f;
    acc += hv * W2[j];
  }
  out[i] = acc;
}

// ---------------- launch ----------------
extern "C" void kernel_launch(void* const* d_in, const int* in_sizes, int n_in,
                              void* d_out, int out_size, void* d_ws, size_t ws_size,
                              hipStream_t stream) {
  (void)in_sizes; (void)n_in; (void)out_size; (void)ws_size;
  const float* x        = (const float*)d_in[0];
  const int*   ei       = (const int*)  d_in[1];
  const float* ea       = (const float*)d_in[2];
  const int*   drug_id  = (const int*)  d_in[3];
  const int*   batch    = (const int*)  d_in[4];
  const float* drug_emb = (const float*)d_in[5];
  const float* Wd       = (const float*)d_in[6];
  const float* bd       = (const float*)d_in[7];
  const float* pW1      = (const float*)d_in[26];
  const float* pb1      = (const float*)d_in[27];
  const float* pW2      = (const float*)d_in[28];
  const float* pb2      = (const float*)d_in[29];

  char* ws = (char*)d_ws;
  float*          df    = (float*)         (ws + OFF_DF);
  float*          xc    = (float*)         (ws + OFF_XC);
  unsigned short* xb    = (unsigned short*)(ws + OFF_XB);
  unsigned short* wt    = (unsigned short*)(ws + OFF_WT);
  float*          Hm    = (float*)         (ws + OFF_H);
  float*          outb  = (float*)         (ws + OFF_OUT);
  float*          asb   = (float*)         (ws + OFF_AS);
  float*          adb   = (float*)         (ws + OFF_AD);
  float*          mb    = (float*)         (ws + OFF_M);
  float*          sb    = (float*)         (ws + OFF_S);
  float*          meanb = (float*)         (ws + OFF_MEAN);
  float*          qb    = (float*)         (ws + OFF_Q);

  // --- drug conditioning + input conversion ---
  drug_df <<<cdiv(NG * HIDC, 256), 256, 0, stream>>>(drug_emb, drug_id, Wd, bd, df);
  add_cond<<<cdiv(NN * HIDC, 256), 256, 0, stream>>>(x, df, batch, xc);
  cvt_bf16<<<cdiv(NN * HIDC, 256), 256, 0, stream>>>(xc, xb, NN * HIDC);
  fill_u32<<<1, 32, 0, stream>>>((uint32_t*)meanb, 0u, 2);
  ea_accum<<<cdiv(NE, 256), 256, 0, stream>>>(ea, meanb);

  struct Layer { const float *W, *We, *as, *ad, *ae, *b; int din, dout, heads, elu; };
  Layer L[3] = {
    { (const float*)d_in[8],  (const float*)d_in[9],  (const float*)d_in[10],
      (const float*)d_in[11], (const float*)d_in[12], (const float*)d_in[13],  64, 256, 4, 1 },
    { (const float*)d_in[14], (const float*)d_in[15], (const float*)d_in[16],
      (const float*)d_in[17], (const float*)d_in[18], (const float*)d_in[19], 256, 256, 4, 1 },
    { (const float*)d_in[20], (const float*)d_in[21], (const float*)d_in[22],
      (const float*)d_in[23], (const float*)d_in[24], (const float*)d_in[25], 256,  64, 1, 0 },
  };

  for (int l = 0; l < 3; ++l) {
    const Layer& Y = L[l];
    // weight transpose -> bf16 [dout, din]
    w_transpose_bf16<<<cdiv(Y.din * Y.dout, 256), 256, 0, stream>>>(Y.W, wt, Y.din, Y.dout);
    // H = Xb @ W  via WMMA bf16 (f32 accumulate), 16x64 strips per wave
    int numTiles = (NN / 16) * (Y.dout / 64);
    gemm_bf16<<<cdiv(numTiles, 4), 128, 0, stream>>>(xb, wt, Hm, NN, Y.dout, Y.din, numTiles);
    // edge-attr attention coefficients
    qcoef<<<1, 32, 0, stream>>>(Y.We, Y.ae, meanb, qb, Y.heads, Y.dout);
    // per-node src/dst attention scalars
    attn_node<<<cdiv(NN * Y.heads, 256), 256, 0, stream>>>(Hm, Y.as, Y.ad, asb, adb, Y.heads, Y.dout);
    // init segment buffers
    fill_u32<<<cdiv(NN * Y.heads, 256), 256, 0, stream>>>((uint32_t*)mb, 0xFF800000u, NN * Y.heads);
    fill_u32<<<cdiv(NN * Y.heads, 256), 256, 0, stream>>>((uint32_t*)sb, 0u, NN * Y.heads);
    fill_u32<<<cdiv(NN * Y.dout, 256), 256, 0, stream>>>((uint32_t*)outb, 0u, NN * Y.dout);
    // segment softmax + weighted scatter (all L2-resident atomics)
    edge_max    <<<cdiv(NET * Y.heads, 256), 256, 0, stream>>>(ei, ea, qb, asb, adb, mb, Y.heads);
    edge_sum    <<<cdiv(NET * Y.heads, 256), 256, 0, stream>>>(ei, ea, qb, asb, adb, mb, sb, Y.heads);
    edge_scatter<<<cdiv(NET, 4), 128, 0, stream>>>(ei, ea, qb, asb, adb, mb, sb, Hm, outb,
                                                   Y.heads, Y.dout);
    // bias + (elu) + bf16 for next layer
    epilogue<<<cdiv(NN * Y.dout, 256), 256, 0, stream>>>(outb, Y.b, xb, Y.dout, Y.elu, NN * Y.dout);
  }

  // concentration head
  head_k<<<cdiv(NN, 256), 256, 0, stream>>>(outb, pW1, pb1, pW2, pb2, (float*)d_out);
}
